// BitsPredictor_65068754534425
// MI455X (gfx1250) — matrix operand order repaired and checked by reading, not compiled, not run on hardware
//
#include <hip/hip_runtime.h>

typedef __bf16 bf16_t;
typedef __attribute__((ext_vector_type(16))) __bf16 v16bf;
typedef __attribute__((ext_vector_type(8)))  __bf16 v8bf;
typedef __attribute__((ext_vector_type(8)))  float  v8f;

#define Bsz 256
#define INsz 2048
#define Ssz 1024
#define Tsz 64
#define Vsz 256
#define G4 4096

// ---------------- WMMA helpers (CDNA5 16x16x32 bf16) ----------------

__device__ __forceinline__ v8f wmma_bf16(v16bf a, v16bf b, v8f c) {
  // (neg_a, A, neg_b, B, c_mod, C, reuse_a, reuse_b)
  return __builtin_amdgcn_wmma_f32_16x16x32_bf16(false, a, false, b, (short)0, c, false, false);
}

// A fragment: 16x32 bf16, A row-major [*, lda].
// lane: m = lane&15, half = lane>>4. elem i<8 -> K = k0+8*half+i ; i>=8 -> K = k0+16+8*half+(i-8)
__device__ __forceinline__ v16bf load_a_frag(const bf16_t* __restrict__ A, int lda,
                                             int row0, int k0, int lane) {
  int m = lane & 15, half = lane >> 4;
  const bf16_t* p = A + (size_t)(row0 + m) * lda + k0 + half * 8;
  v8bf lo = *(const v8bf*)p;
  v8bf hi = *(const v8bf*)(p + 16);
  v16bf r;
#pragma unroll
  for (int i = 0; i < 8; ++i) { r[i] = lo[i]; r[i + 8] = hi[i]; }
  return r;
}

// B fragment: 32x16 bf16 from weights stored N-major: Wt[n][k], ld = K.
// lane: n = lane&15, half = lane>>4. elem i<8 -> K = k0+8*half+i ; i>=8 -> K = k0+16+8*half+(i-8)
__device__ __forceinline__ v16bf load_b_frag(const bf16_t* __restrict__ Wt, int ldb,
                                             int n0, int k0, int lane) {
  int n = lane & 15, half = lane >> 4;
  const bf16_t* p = Wt + (size_t)(n0 + n) * ldb + k0 + half * 8;
  v8bf lo = *(const v8bf*)p;
  v8bf hi = *(const v8bf*)(p + 16);
  v16bf r;
#pragma unroll
  for (int i = 0; i < 8; ++i) { r[i] = lo[i]; r[i + 8] = hi[i]; }
  return r;
}

// ---------------- Generic WMMA GEMM: C[M,N] = A[M,K] @ Wt^T + bias ----------------
// Block: 256 threads = 8 waves (4x2), block tile 128(M) x 64(N), wave tile 32x32.
template <bool OUT_BF16, bool PERMUTE>
__global__ __launch_bounds__(256) void gemm_wmma_kernel(
    const bf16_t* __restrict__ A, const bf16_t* __restrict__ Wt,
    const float* __restrict__ bias, void* __restrict__ out,
    int M, int N, int K, int Bp, int Tp) {
  int lane = threadIdx.x & 31;
  int wave = threadIdx.x >> 5;
  int m0 = blockIdx.x * 128 + (wave >> 1) * 32;
  int n0 = blockIdx.y * 64 + (wave & 1) * 32;
  v8f acc[2][2] = {};
  for (int k0 = 0; k0 < K; k0 += 32) {
    if (k0 + 32 < K) {
      __builtin_prefetch(A + (size_t)(m0 + (lane & 15)) * K + k0 + 32, 0, 1);
      __builtin_prefetch(Wt + (size_t)(n0 + (lane & 15)) * K + k0 + 32, 0, 1);
    }
    v16bf a0 = load_a_frag(A, K, m0, k0, lane);
    v16bf a1 = load_a_frag(A, K, m0 + 16, k0, lane);
    v16bf b0 = load_b_frag(Wt, K, n0, k0, lane);
    v16bf b1 = load_b_frag(Wt, K, n0 + 16, k0, lane);
    acc[0][0] = wmma_bf16(a0, b0, acc[0][0]);
    acc[0][1] = wmma_bf16(a0, b1, acc[0][1]);
    acc[1][0] = wmma_bf16(a1, b0, acc[1][0]);
    acc[1][1] = wmma_bf16(a1, b1, acc[1][1]);
  }
  int nn = lane & 15, half = lane >> 4;
#pragma unroll
  for (int ti = 0; ti < 2; ++ti)
#pragma unroll
    for (int tj = 0; tj < 2; ++tj)
#pragma unroll
      for (int r = 0; r < 8; ++r) {
        int row = m0 + ti * 16 + half * 8 + r;
        int col = n0 + tj * 16 + nn;
        float v = acc[ti][tj][r];
        if (bias) v += bias[col];
        size_t orow = PERMUTE ? (size_t)(row % Bp) * Tp + (row / Bp) : (size_t)row;
        if (OUT_BF16) ((bf16_t*)out)[orow * N + col] = (bf16_t)v;
        else          ((float*) out)[orow * N + col] = v;
      }
}

// ---------------- Fused LSTM step: gates = xproj[t] + h @ Whh ; cell update ------
// Grid: (B/32, S/64). 8 waves: wave -> gate group g = wave>>1 (i,f,g,o), col half = wave&1.
__device__ __forceinline__ float sigf(float x) { return 1.0f / (1.0f + expf(-x)); }

__global__ __launch_bounds__(256) void lstm_step_kernel(
    const bf16_t* __restrict__ h_in, bf16_t* __restrict__ h_out,
    float* __restrict__ c, const bf16_t* __restrict__ Whh_t,
    const bf16_t* __restrict__ xprojt, bf16_t* __restrict__ out_t) {
  __shared__ float gsm[4][32][64];  // 32 KB
  int lane = threadIdx.x & 31;
  int wave = threadIdx.x >> 5;
  int g = wave >> 1, ch = wave & 1;
  int m0 = blockIdx.x * 32;
  int s0 = blockIdx.y * 64;
  int ng0 = g * Ssz + s0 + ch * 32;  // gate column base in [0,4096)
  v8f acc[2][2] = {};
  for (int k0 = 0; k0 < Ssz; k0 += 32) {
    v16bf a0 = load_a_frag(h_in, Ssz, m0, k0, lane);
    v16bf a1 = load_a_frag(h_in, Ssz, m0 + 16, k0, lane);
    v16bf b0 = load_b_frag(Whh_t, Ssz, ng0, k0, lane);
    v16bf b1 = load_b_frag(Whh_t, Ssz, ng0 + 16, k0, lane);
    acc[0][0] = wmma_bf16(a0, b0, acc[0][0]);
    acc[0][1] = wmma_bf16(a0, b1, acc[0][1]);
    acc[1][0] = wmma_bf16(a1, b0, acc[1][0]);
    acc[1][1] = wmma_bf16(a1, b1, acc[1][1]);
  }
  int nn = lane & 15, half = lane >> 4;
#pragma unroll
  for (int ti = 0; ti < 2; ++ti)
#pragma unroll
    for (int tj = 0; tj < 2; ++tj)
#pragma unroll
      for (int r = 0; r < 8; ++r) {
        int rl = ti * 16 + half * 8 + r;       // 0..31 row in tile
        int cg = ng0 + tj * 16 + nn;           // global gate col
        int cl = ch * 32 + tj * 16 + nn;       // 0..63 state col in tile
        gsm[g][rl][cl] = acc[ti][tj][r] + (float)xprojt[(size_t)(m0 + rl) * G4 + cg];
      }
  __syncthreads();
#pragma unroll
  for (int e = 0; e < 8; ++e) {
    int idx = threadIdx.x + e * 256;           // 0..2047 over 32x64
    int rl = idx >> 6, cl = idx & 63;
    float gi = gsm[0][rl][cl], gf = gsm[1][rl][cl];
    float gg = gsm[2][rl][cl], go = gsm[3][rl][cl];
    size_t off = (size_t)(m0 + rl) * Ssz + s0 + cl;
    float cp = c[off];
    float cn = sigf(gf) * cp + sigf(gi) * tanhf(gg);
    float hn = sigf(go) * tanhf(cn);
    c[off] = cn;
    bf16_t hb = (bf16_t)hn;
    h_out[off] = hb;
    out_t[off] = hb;
  }
}

// ---------------- Conversion / prep kernels ----------------

__global__ void cvt_f32_bf16_kernel(const float* __restrict__ src, bf16_t* __restrict__ dst, int n) {
  int i = blockIdx.x * blockDim.x + threadIdx.x;
  if (i < n) dst[i] = (bf16_t)src[i];
}

// dst[n][k] = (bf16) src[k][n] ; src is [K][N] row-major. Grid (K/32, N/32), 256 thr.
__global__ void transpose_cvt_kernel(const float* __restrict__ src, bf16_t* __restrict__ dst,
                                     int K, int N) {
  __shared__ float tile[32][33];
  int k0 = blockIdx.x * 32, n0 = blockIdx.y * 32;
  int tx = threadIdx.x & 31, ty = threadIdx.x >> 5;  // ty 0..7
#pragma unroll
  for (int j = 0; j < 32; j += 8)
    tile[ty + j][tx] = src[(size_t)(k0 + ty + j) * N + n0 + tx];
  __syncthreads();
#pragma unroll
  for (int j = 0; j < 32; j += 8)
    dst[(size_t)(n0 + ty + j) * K + k0 + tx] = (bf16_t)tile[tx][ty + j];
}

__global__ void bias_cat3_kernel(const float* b1, const float* b2, const float* b3, float* bc) {
  int i = blockIdx.x * blockDim.x + threadIdx.x;
  if (i < Ssz) { bc[i] = b1[i]; bc[Ssz + i] = b2[i]; bc[2 * Ssz + i] = b3[i]; }
}

__global__ void bias_add_kernel(const float* a, const float* b, float* o, int n) {
  int i = blockIdx.x * blockDim.x + threadIdx.x;
  if (i < n) o[i] = a[i] + b[i];
}

__global__ void pack_bits_kernel(const int* __restrict__ bits, int* __restrict__ tgt) {
  int i = blockIdx.x * blockDim.x + threadIdx.x;  // over B*T
  if (i >= Bsz * Tsz) return;
  int b = i / Tsz, t = i % Tsz;
  const int* p = bits + b * (Tsz * 8) + t * 8;
  int v = 0;
#pragma unroll
  for (int j = 0; j < 8; ++j) v |= ((p[j] > 0) ? 1 : 0) << j;  // LSB-first
  tgt[i] = v;
}

// fhc [B][3072] -> teacher[0] (bf16), h0 (bf16), c0 (f32)
__global__ void split_fhc_kernel(const float* __restrict__ fhc, bf16_t* __restrict__ teacher0,
                                 bf16_t* __restrict__ h0, float* __restrict__ c0) {
  int i = blockIdx.x * blockDim.x + threadIdx.x;  // over B*S
  if (i >= Bsz * Ssz) return;
  int b = i >> 10, s = i & (Ssz - 1);
  const float* row = fhc + (size_t)b * (3 * Ssz);
  teacher0[i] = (bf16_t)row[s];
  h0[i] = (bf16_t)row[Ssz + s];
  c0[i] = row[2 * Ssz + s];
}

// teacher[t][b][s] = bf16( W4[tgt[b][t-1]][s] + b4[s] ), t = 1..63
__global__ void gather_teacher_kernel(const float* __restrict__ W4, const float* __restrict__ b4,
                                      const int* __restrict__ tgt, bf16_t* __restrict__ teacher) {
  size_t i = (size_t)blockIdx.x * blockDim.x + threadIdx.x;  // over (T-1)*B*S
  if (i >= (size_t)(Tsz - 1) * Bsz * Ssz) return;
  int s = (int)(i & (Ssz - 1));
  int bt = (int)(i >> 10);
  int b = bt & (Bsz - 1);
  int tm1 = bt >> 8;  // 0..62
  int ti = tgt[b * Tsz + tm1];
  teacher[((size_t)(tm1 + 1) * Bsz + b) * Ssz + s] = (bf16_t)(W4[(size_t)ti * Ssz + s] + b4[s]);
}

__global__ void zero_kernel(float* p) { *p = 0.0f; }

// one block (256 thr) per (b,t) row of pred; mean NLL accumulated into *loss
__global__ __launch_bounds__(256) void nll_kernel(const float* __restrict__ pred,
                                                  const int* __restrict__ tgt,
                                                  float* __restrict__ loss) {
  __shared__ float red[256];
  int r = blockIdx.x;            // row in [B*T), layout b*T + t
  int tid = threadIdx.x;
  float v = pred[(size_t)r * Vsz + tid];
  red[tid] = v;
  __syncthreads();
  for (int s = 128; s > 0; s >>= 1) {
    if (tid < s) red[tid] = fmaxf(red[tid], red[tid + s]);
    __syncthreads();
  }
  float mx = red[0];
  __syncthreads();
  red[tid] = expf(v - mx);
  __syncthreads();
  for (int s = 128; s > 0; s >>= 1) {
    if (tid < s) red[tid] += red[tid + s];
    __syncthreads();
  }
  if (tid == 0) {
    int tg = tgt[r];  // tgt is [B][T], r = b*T+t matches
    float nll = logf(red[0]) + mx - pred[(size_t)r * Vsz + tg];
    atomicAdd(loss, nll * (1.0f / (float)(Bsz * Tsz)));
  }
}

// ---------------- Host-side orchestration ----------------

extern "C" void kernel_launch(void* const* d_in, const int* in_sizes, int n_in,
                              void* d_out, int out_size, void* d_ws, size_t ws_size,
                              hipStream_t stream) {
  const float* x  = (const float*)d_in[0];
  const int* tbits = (const int*)d_in[1];
  // d_in[2] = temperature (unused on teacher-forcing path)
  const float* W1 = (const float*)d_in[3];  const float* b1 = (const float*)d_in[4];
  const float* W2 = (const float*)d_in[5];  const float* b2 = (const float*)d_in[6];
  const float* W3 = (const float*)d_in[7];  const float* b3 = (const float*)d_in[8];
  const float* W4 = (const float*)d_in[9];  const float* b4 = (const float*)d_in[10];
  const float* W5 = (const float*)d_in[11]; const float* b5 = (const float*)d_in[12];
  const float* Wih = (const float*)d_in[13]; const float* Whh = (const float*)d_in[14];
  const float* bih = (const float*)d_in[15]; const float* bhh = (const float*)d_in[16];

  char* ws = (char*)d_ws;
  size_t off = 0;
  auto take = [&](size_t bytes) -> char* {
    char* p = ws + off;
    off += (bytes + 255) & ~(size_t)255;
    return p;
  };
  bf16_t* xb     = (bf16_t*)take((size_t)Bsz * INsz * 2);
  bf16_t* Wc_t   = (bf16_t*)take((size_t)3 * Ssz * INsz * 2);   // [3072][2048]
  float*  bc     = (float*) take((size_t)3 * Ssz * 4);
  bf16_t* Wih_t  = (bf16_t*)take((size_t)G4 * Ssz * 2);         // [4096][1024]
  bf16_t* Whh_t  = (bf16_t*)take((size_t)G4 * Ssz * 2);
  float*  bg     = (float*) take((size_t)G4 * 4);
  bf16_t* W5t    = (bf16_t*)take((size_t)Vsz * Ssz * 2);        // [256][1024]
  float*  fhc    = (float*) take((size_t)Bsz * 3 * Ssz * 4);
  bf16_t* teach  = (bf16_t*)take((size_t)Tsz * Bsz * Ssz * 2);  // [T][B][S]
  int*    tgt    = (int*)   take((size_t)Bsz * Tsz * 4);
  bf16_t* xproj  = (bf16_t*)take((size_t)Tsz * Bsz * G4 * 2);   // [T][B][4096]
  bf16_t* hbuf0  = (bf16_t*)take((size_t)Bsz * Ssz * 2);
  bf16_t* hbuf1  = (bf16_t*)take((size_t)Bsz * Ssz * 2);
  float*  cbuf   = (float*) take((size_t)Bsz * Ssz * 4);
  bf16_t* outs   = (bf16_t*)take((size_t)Tsz * Bsz * Ssz * 2);  // [T][B][S]

  float* pred = (float*)d_out;                    // [B][T][256]
  float* loss = (float*)d_out + (out_size - 1);   // scalar

  // ---- weight conversion (f32 -> bf16, N-major) ----
  cvt_f32_bf16_kernel<<<(Bsz * INsz + 255) / 256, 256, 0, stream>>>(x, xb, Bsz * INsz);
  transpose_cvt_kernel<<<dim3(INsz / 32, Ssz / 32), 256, 0, stream>>>(W1, Wc_t,                 INsz, Ssz);
  transpose_cvt_kernel<<<dim3(INsz / 32, Ssz / 32), 256, 0, stream>>>(W2, Wc_t + (size_t)Ssz * INsz,     INsz, Ssz);
  transpose_cvt_kernel<<<dim3(INsz / 32, Ssz / 32), 256, 0, stream>>>(W3, Wc_t + (size_t)2 * Ssz * INsz, INsz, Ssz);
  transpose_cvt_kernel<<<dim3(Ssz / 32, G4 / 32), 256, 0, stream>>>(Wih, Wih_t, Ssz, G4);
  transpose_cvt_kernel<<<dim3(Ssz / 32, G4 / 32), 256, 0, stream>>>(Whh, Whh_t, Ssz, G4);
  transpose_cvt_kernel<<<dim3(Ssz / 32, Vsz / 32), 256, 0, stream>>>(W5, W5t, Ssz, Vsz);
  bias_cat3_kernel<<<4, 256, 0, stream>>>(b1, b2, b3, bc);
  bias_add_kernel<<<G4 / 256, 256, 0, stream>>>(bih, bhh, bg, G4);

  // ---- target ints ----
  pack_bits_kernel<<<(Bsz * Tsz + 255) / 256, 256, 0, stream>>>(tbits, tgt);

  // ---- GEMM1: [first|h0|c0] = x @ [W1|W2|W3] + [b1|b2|b3] ----
  gemm_wmma_kernel<false, false><<<dim3(Bsz / 128, 3 * Ssz / 64), 256, 0, stream>>>(
      xb, Wc_t, bc, fhc, Bsz, 3 * Ssz, INsz, 0, 0);
  split_fhc_kernel<<<(Bsz * Ssz + 255) / 256, 256, 0, stream>>>(fhc, teach, hbuf0, cbuf);

  // ---- teacher embedding gather for t=1..63 ----
  gather_teacher_kernel<<<(unsigned)(((size_t)(Tsz - 1) * Bsz * Ssz + 255) / 256), 256, 0, stream>>>(
      W4, b4, tgt, teach);

  // ---- GEMM2: xproj = teacher @ Wih + (bih+bhh)  (one big GEMM, all T steps) ----
  gemm_wmma_kernel<true, false><<<dim3(Tsz * Bsz / 128, G4 / 64), 256, 0, stream>>>(
      teach, Wih_t, bg, xproj, Tsz * Bsz, G4, Ssz, 0, 0);

  // ---- recurrent loop: 64 fused (h@Whh + cell) steps ----
  for (int t = 0; t < Tsz; ++t) {
    const bf16_t* hin = (t & 1) ? hbuf1 : hbuf0;
    bf16_t* hout = (t & 1) ? hbuf0 : hbuf1;
    lstm_step_kernel<<<dim3(Bsz / 32, Ssz / 64), 256, 0, stream>>>(
        hin, hout, cbuf, Whh_t,
        xproj + (size_t)t * Bsz * G4,
        outs + (size_t)t * Bsz * Ssz);
  }

  // ---- GEMM3: pred[b][t][:] = outputs @ W5 + b5, with [T,B]->[B,T] row permute ----
  gemm_wmma_kernel<false, true><<<dim3(Tsz * Bsz / 128, Vsz / 64), 256, 0, stream>>>(
      outs, W5t, b5, pred, Tsz * Bsz, Vsz, Ssz, Bsz, Tsz);

  // ---- loss: mean NLL over [B,T] ----
  zero_kernel<<<1, 1, 0, stream>>>(loss);
  nll_kernel<<<Bsz * Tsz, 256, 0, stream>>>(pred, tgt, loss);
}
